// TransitionUp_85461259256091
// MI455X (gfx1250) — matrix operand order repaired and checked by reading, not compiled, not run on hardware
//
#include <hip/hip_runtime.h>
#include <stdint.h>
#include <stddef.h>

typedef __attribute__((ext_vector_type(16))) __bf16 v16bf;
typedef __attribute__((ext_vector_type(8)))  float  v8f;

#define EPS_W   1e-8f
#define BN_EPS  1e-5f
#define BATCH   4
#define NPTS1   8192
#define NPTS2   2048
#define CIN     512
#define COUT    256

// ---------------------------------------------------------------- helpers
static __device__ __forceinline__ uint16_t f32_to_bf16_rne(float f) {
    union { float f; uint32_t u; } x; x.f = f;
    uint32_t u = x.u;
    uint32_t r = u + 0x7FFFu + ((u >> 16) & 1u);   // round-to-nearest-even
    return (uint16_t)(r >> 16);
}

// ---------------------------------------------------------------- f32 -> bf16 pack
__global__ __launch_bounds__(256)
void cvt_bf16_kernel(const float* __restrict__ in, uint16_t* __restrict__ out, int n) {
    int i = blockIdx.x * blockDim.x + threadIdx.x;
    if (i < n) out[i] = f32_to_bf16_rne(in[i]);
}

// ---------------------------------------------------------------- WMMA GEMM + bias
// out[M,N] = A[M,K](bf16) * W[N,K](bf16)^T + bias[N]
// 128 threads = 4 waves; wave computes 16(M) x 64(N); block = 64x64 tile.
__global__ __launch_bounds__(128)
void gemm_bias_wmma_kernel(const uint16_t* __restrict__ Abf,
                           const uint16_t* __restrict__ Wbf,
                           const float* __restrict__ bias,
                           float* __restrict__ out,
                           int M, int N, int K) {
    const uint32_t* A32 = (const uint32_t*)Abf;   // 2 bf16 per dword
    const uint32_t* W32 = (const uint32_t*)Wbf;
    const int K2 = K >> 1;

    const int lane = threadIdx.x & 31;
    const int wave = threadIdx.x >> 5;
    const int half = lane >> 4;     // lane group: K-offset selector
    const int l16  = lane & 15;

    const int m0 = blockIdx.x * 64 + wave * 16;
    const int n0 = blockIdx.y * 64;

    v8f acc[4] = {v8f{}, v8f{}, v8f{}, v8f{}};

    // A fragment base: row = m0+l16 (lanes 0-15 and 16-31 carry same rows,
    // different K halves: k = 16*half + s, same order as B).
    const int row = m0 + l16;
    const uint32_t* arow  = A32 + (size_t)row * K2 + 8 * half;
    // B fragment base: col = n0 + nt*16 + l16, K pairs i + 8*half: contiguous
    // k = 16*half + s, same order as A.
    const uint32_t* wbase = W32 + (size_t)(n0 + l16) * K2 + 8 * half;

    const int ksteps = K >> 5;     // 32 K-elements per WMMA
    for (int kt = 0; kt < ksteps; ++kt) {
        union { uint32_t u[8]; v16bf v; } af;
        const uint32_t* ak = arow + kt * 16;
        af.u[0] = ak[0];  af.u[1] = ak[1];  af.u[2] = ak[2];  af.u[3] = ak[3];
        af.u[4] = ak[4];  af.u[5] = ak[5];  af.u[6] = ak[6];  af.u[7] = ak[7];
        __builtin_prefetch(ak + 16, 0, 3);    // next K tile -> global_prefetch

        #pragma unroll
        for (int nt = 0; nt < 4; ++nt) {
            union { uint32_t u[8]; v16bf v; } bf;
            const uint32_t* wk = wbase + (size_t)nt * 16 * K2 + kt * 16;
            #pragma unroll
            for (int i = 0; i < 8; ++i) bf.u[i] = wk[i];
            acc[nt] = __builtin_amdgcn_wmma_f32_16x16x32_bf16(
                false, af.v, false, bf.v, (short)0, acc[nt], false, false);
        }
    }

    // C/D layout: lanes 0-15 -> M = i, lanes 16-31 -> M = 8+i; N = l16.
    #pragma unroll
    for (int nt = 0; nt < 4; ++nt) {
        const int c = n0 + nt * 16 + l16;
        const float bc = bias[c];
        #pragma unroll
        for (int i = 0; i < 8; ++i) {
            const int r = m0 + i + 8 * half;
            out[(size_t)r * N + c] = acc[nt][i] + bc;
        }
    }
}

// ---------------------------------------------------------------- BN stats (deterministic 2-stage)
__global__ __launch_bounds__(256)
void bn_stats_kernel(const float* __restrict__ h, int rows_per_block,
                     float* __restrict__ psum, float* __restrict__ psq) {
    const int c  = threadIdx.x;                 // channel; consecutive -> coalesced
    const int r0 = blockIdx.x * rows_per_block;
    float s = 0.f, q = 0.f;
    for (int r = r0; r < r0 + rows_per_block; ++r) {
        float v = h[(size_t)r * COUT + c];
        s += v; q += v * v;
    }
    psum[(size_t)blockIdx.x * COUT + c] = s;
    psq [(size_t)blockIdx.x * COUT + c] = q;
}

__global__ __launch_bounds__(256)
void bn_finalize_kernel(const float* __restrict__ psum, const float* __restrict__ psq,
                        int nblocks, int M,
                        const float* __restrict__ gamma, const float* __restrict__ beta,
                        float* __restrict__ scale, float* __restrict__ shift) {
    const int c = threadIdx.x;
    float s = 0.f, q = 0.f;
    for (int b = 0; b < nblocks; ++b) {
        s += psum[(size_t)b * COUT + c];
        q += psq [(size_t)b * COUT + c];
    }
    const float mean = s / (float)M;
    const float var  = q / (float)M - mean * mean;   // biased, matches torch BN
    const float inv  = rsqrtf(var + BN_EPS);
    const float sc   = inv * gamma[c];
    scale[c] = sc;
    shift[c] = beta[c] - mean * sc;
}

__global__ __launch_bounds__(256)
void bn_relu_apply_kernel(float* __restrict__ h, const float* __restrict__ scale,
                          const float* __restrict__ shift, size_t n) {
    size_t i = (size_t)blockIdx.x * blockDim.x + threadIdx.x;
    if (i < n) {
        int c = (int)(i & (COUT - 1));
        float v = h[i] * scale[c] + shift[c];
        h[i] = v > 0.f ? v : 0.f;
    }
}

// ---------------------------------------------------------------- 3-NN with points2 staged in LDS
__global__ __launch_bounds__(256)
void knn3_kernel(const float* __restrict__ points1, const float* __restrict__ points2,
                 int* __restrict__ kidx, float* __restrict__ kw) {
    __shared__ float sx[NPTS2], sy[NPTS2], sz[NPTS2];   // 24 KB of LDS
    const int b = blockIdx.y;
    const float* p2 = points2 + (size_t)b * NPTS2 * 3;
    for (int j = threadIdx.x; j < NPTS2; j += blockDim.x) {
        sx[j] = p2[j * 3 + 0];
        sy[j] = p2[j * 3 + 1];
        sz[j] = p2[j * 3 + 2];
    }
    __syncthreads();

    const int n = blockIdx.x * blockDim.x + threadIdx.x;      // query in [0, NPTS1)
    const float* p1 = points1 + ((size_t)b * NPTS1 + n) * 3;
    const float px = p1[0], py = p1[1], pz = p1[2];

    float d0 = 1e30f, d1 = 1e30f, d2 = 1e30f;
    int   i0 = 0,     i1 = 0,     i2 = 0;
    for (int j = 0; j < NPTS2; ++j) {
        const float dx = px - sx[j], dy = py - sy[j], dz = pz - sz[j];
        const float d  = dx * dx + dy * dy + dz * dz;
        if (d < d0)      { d2 = d1; i2 = i1; d1 = d0; i1 = i0; d0 = d; i0 = j; }
        else if (d < d1) { d2 = d1; i2 = i1; d1 = d;  i1 = j; }
        else if (d < d2) { d2 = d;  i2 = j; }
    }
    float w0 = 1.f / (sqrtf(fmaxf(d0, 0.f)) + EPS_W);
    float w1 = 1.f / (sqrtf(fmaxf(d1, 0.f)) + EPS_W);
    float w2 = 1.f / (sqrtf(fmaxf(d2, 0.f)) + EPS_W);
    const float wsum = w0 + w1 + w2;
    const size_t q = (size_t)b * NPTS1 + n;
    kw[q * 3 + 0] = w0 / wsum;
    kw[q * 3 + 1] = w1 / wsum;
    kw[q * 3 + 2] = w2 / wsum;
    kidx[q * 3 + 0] = b * NPTS2 + i0;
    kidx[q * 3 + 1] = b * NPTS2 + i1;
    kidx[q * 3 + 2] = b * NPTS2 + i2;
}

// ---------------------------------------------------------------- gather + weighted sum + add
__global__ __launch_bounds__(256)
void interp_add_kernel(const float* __restrict__ h1, const float* __restrict__ h2,
                       const int* __restrict__ kidx, const float* __restrict__ kw,
                       float* __restrict__ out) {
    const size_t q = blockIdx.x;
    const int    c = threadIdx.x;
    const int j0 = kidx[q * 3 + 0], j1 = kidx[q * 3 + 1], j2 = kidx[q * 3 + 2];
    const float w0 = kw[q * 3 + 0], w1 = kw[q * 3 + 1], w2 = kw[q * 3 + 2];
    const float v = h1[q * COUT + c]
                  + w0 * h2[(size_t)j0 * COUT + c]
                  + w1 * h2[(size_t)j1 * COUT + c]
                  + w2 * h2[(size_t)j2 * COUT + c];
    out[q * COUT + c] = v;
}

__global__ __launch_bounds__(256)
void copy_f32_kernel(const float* __restrict__ in, float* __restrict__ out, int n) {
    int i = blockIdx.x * blockDim.x + threadIdx.x;
    if (i < n) out[i] = in[i];
}

// ---------------------------------------------------------------- host launch
extern "C" void kernel_launch(void* const* d_in, const int* in_sizes, int n_in,
                              void* d_out, int out_size, void* d_ws, size_t ws_size,
                              hipStream_t stream) {
    (void)in_sizes; (void)n_in; (void)out_size; (void)ws_size;

    const float* feats1  = (const float*)d_in[0];   // [4,8192,256]
    const float* points1 = (const float*)d_in[1];   // [4,8192,3]
    const float* feats2  = (const float*)d_in[2];   // [4,2048,512]
    const float* points2 = (const float*)d_in[3];   // [4,2048,3]
    const float* f1_w    = (const float*)d_in[4];   // [256,256]
    const float* f1_b    = (const float*)d_in[5];
    const float* f1_g    = (const float*)d_in[6];
    const float* f1_be   = (const float*)d_in[7];
    const float* f2_w    = (const float*)d_in[8];   // [256,512]
    const float* f2_b    = (const float*)d_in[9];
    const float* f2_g    = (const float*)d_in[10];
    const float* f2_be   = (const float*)d_in[11];

    const int M1 = BATCH * NPTS1;   // 32768
    const int M2 = BATCH * NPTS2;   // 8192
    const int K1 = COUT;            // feats1 channel count = 256
    const int K2 = CIN;             // 512

    // workspace layout
    char* ws = (char*)d_ws;
    size_t off = 0;
    auto alloc = [&](size_t bytes) {
        void* p = ws + off;
        off = (off + bytes + 255) & ~(size_t)255;
        return p;
    };
    uint16_t* a1_bf  = (uint16_t*)alloc((size_t)M1 * K1 * 2);
    uint16_t* a2_bf  = (uint16_t*)alloc((size_t)M2 * K2 * 2);
    uint16_t* w1_bf  = (uint16_t*)alloc((size_t)COUT * K1 * 2);
    uint16_t* w2_bf  = (uint16_t*)alloc((size_t)COUT * K2 * 2);
    float*    h1     = (float*)alloc((size_t)M1 * COUT * 4);
    float*    h2     = (float*)alloc((size_t)M2 * COUT * 4);
    float*    psum   = (float*)alloc((size_t)256 * COUT * 4);
    float*    psq    = (float*)alloc((size_t)256 * COUT * 4);
    float*    scale1 = (float*)alloc(COUT * 4);
    float*    shift1 = (float*)alloc(COUT * 4);
    float*    scale2 = (float*)alloc(COUT * 4);
    float*    shift2 = (float*)alloc(COUT * 4);
    int*      kidx   = (int*)alloc((size_t)M1 * 3 * 4);
    float*    kw     = (float*)alloc((size_t)M1 * 3 * 4);

    // 1) f32 -> bf16 packing of activations and weights
    cvt_bf16_kernel<<<(M1 * K1 + 255) / 256, 256, 0, stream>>>(feats1, a1_bf, M1 * K1);
    cvt_bf16_kernel<<<(M2 * K2 + 255) / 256, 256, 0, stream>>>(feats2, a2_bf, M2 * K2);
    cvt_bf16_kernel<<<(COUT * K1 + 255) / 256, 256, 0, stream>>>(f1_w, w1_bf, COUT * K1);
    cvt_bf16_kernel<<<(COUT * K2 + 255) / 256, 256, 0, stream>>>(f2_w, w2_bf, COUT * K2);

    // 2) WMMA GEMMs with fused bias
    dim3 g1(M1 / 64, COUT / 64);
    gemm_bias_wmma_kernel<<<g1, 128, 0, stream>>>(a1_bf, w1_bf, f1_b, h1, M1, COUT, K1);
    dim3 g2(M2 / 64, COUT / 64);
    gemm_bias_wmma_kernel<<<g2, 128, 0, stream>>>(a2_bf, w2_bf, f2_b, h2, M2, COUT, K2);

    // 3) BN (training stats) + ReLU, layer 1 (in place on h1)
    bn_stats_kernel<<<256, 256, 0, stream>>>(h1, M1 / 256, psum, psq);
    bn_finalize_kernel<<<1, 256, 0, stream>>>(psum, psq, 256, M1, f1_g, f1_be, scale1, shift1);
    bn_relu_apply_kernel<<<(int)(((size_t)M1 * COUT) / 256), 256, 0, stream>>>(
        h1, scale1, shift1, (size_t)M1 * COUT);

    // 4) BN + ReLU, layer 2 (in place on h2)
    bn_stats_kernel<<<256, 256, 0, stream>>>(h2, M2 / 256, psum, psq);
    bn_finalize_kernel<<<1, 256, 0, stream>>>(psum, psq, 256, M2, f2_g, f2_be, scale2, shift2);
    bn_relu_apply_kernel<<<(int)(((size_t)M2 * COUT) / 256), 256, 0, stream>>>(
        h2, scale2, shift2, (size_t)M2 * COUT);

    // 5) 3-NN (points2 staged in LDS) -> indices + normalized inverse-distance weights
    knn3_kernel<<<dim3(NPTS1 / 256, BATCH), 256, 0, stream>>>(points1, points2, kidx, kw);

    // 6) gather + weighted sum + h1 add -> first output region
    float* out0 = (float*)d_out;
    interp_add_kernel<<<M1, 256, 0, stream>>>(h1, h2, kidx, kw, out0);

    // 7) second tuple output: points1 pass-through
    copy_f32_kernel<<<(BATCH * NPTS1 * 3 + 255) / 256, 256, 0, stream>>>(
        points1, out0 + (size_t)M1 * COUT, BATCH * NPTS1 * 3);
}
